// WordLSTM_369367187773
// MI455X (gfx1250) — compile-verified, matrix operand order checked by reading, not run on hardware
//
#include <hip/hip_runtime.h>
#include <hip/hip_bf16.h>

#define T_SEQ 256
#define B_SZ  32
#define H_SZ  1024
#define L_SZ  2
#define V_SZ  10000
#define G4H   (4 * H_SZ)

typedef __attribute__((ext_vector_type(16))) __bf16 v16bf;
typedef __attribute__((ext_vector_type(8)))  __bf16 v8bf;
typedef __attribute__((ext_vector_type(8)))  float  v8f;

static __device__ inline __bf16 f2bf(float f) {
    union { float f; unsigned u; } v; v.f = f;
    unsigned r = v.u + 0x7FFFu + ((v.u >> 16) & 1u);   // round-to-nearest-even
    unsigned short h = (unsigned short)(r >> 16);
    return __builtin_bit_cast(__bf16, h);
}

// Pack one 16x32 bf16 WMMA operand fragment from a per-lane base pointer.
// Per ISA 7.12.2 (16-bit A 16x32 layout; identical index pattern for B 32x16):
//   lane 0-15  : row = lane,    K = 0-7 (elems 0-7),  K = 16-23 (elems 8-15)
//   lane 16-31 : row = lane-16, K = 8-15 (elems 0-7), K = 24-31 (elems 8-15)
// The caller precomputes base = &M[row_for_lane][klo_for_lane]; this is two
// contiguous 16-byte loads -> a global_load_b128 pair with immediate offsets.
static __device__ inline v16bf frag_at(const __bf16* __restrict__ base) {
    v8bf lo = *(const v8bf*)(base);
    v8bf hi = *(const v8bf*)(base + 16);
    v16bf a;
#pragma unroll
    for (int i = 0; i < 8; ++i) { a[i] = lo[i]; a[i + 8] = hi[i]; }
    return a;
}

static __device__ inline v8f wmma_bf16(v16bf a, v16bf b, v8f c) {
    return __builtin_amdgcn_wmma_f32_16x16x32_bf16(
        /*neg_a=*/false, a, /*neg_b=*/false, b,
        /*c_mod=*/(short)0, c, /*reuse_a=*/false, /*reuse_b=*/false);
}

// ---------------------------------------------------------------------------
// C[M,N] = A[M,K] @ B[N,K]^T + bias0[N] + bias1[N]   (bf16 in, f32 out)
// K is a template parameter so all row strides are compile-time constants:
// the 8 fragment base pointers (4 A rows, 4 B rows, per-lane) are hoisted out
// of the K loop and every load inside is base+k -> immediate-offset b128
// loads with a single pointer bump, no per-iteration VALU address math (which
// previously caused WMMA->VALU WAR hazard v_nop bubbles).
// Wave tile 64x64 = 4x4 f32 accumulators: 16 WMMAs per 16 b128 loads per
// K-step of 32 (1 load per WMMA). Block = 8 waves (2Mx4N) -> tile 128x256.
// CLAMP handles N not a multiple of 16 (decoder): B rows clamp to N-1 (safe:
// WMMA column n depends only on B column n; clamped columns never stored).
// ---------------------------------------------------------------------------
template <int K, bool CLAMP>
__global__ __launch_bounds__(256, 1) void gemm_bf16_nt(
    const __bf16* __restrict__ A, const __bf16* __restrict__ Bw,
    const float* __restrict__ bias0, const float* __restrict__ bias1,
    float* __restrict__ C, int M, int N)
{
    const int wave = threadIdx.x >> 5;
    const int lane = threadIdx.x & 31;
    const int wm = wave >> 2, wn = wave & 3;
    const int m0 = blockIdx.y * 128 + wm * 64;
    const int n0 = blockIdx.x * 256 + wn * 64;

    const int rr  = lane & 15;
    const int klo = (lane >> 4) << 3;

    // Loop-invariant per-lane fragment base pointers.
    const __bf16* Ab[4];
    const __bf16* Bb[4];
#pragma unroll
    for (int i = 0; i < 4; ++i)
        Ab[i] = A + (size_t)(m0 + 16 * i + rr) * K + klo;
#pragma unroll
    for (int j = 0; j < 4; ++j) {
        int row = n0 + 16 * j + rr;
        if (CLAMP && row > N - 1) row = N - 1;
        Bb[j] = Bw + (size_t)row * K + klo;
    }

    v8f acc[4][4];
#pragma unroll
    for (int i = 0; i < 4; ++i)
#pragma unroll
        for (int j = 0; j < 4; ++j) acc[i][j] = (v8f){};

#pragma unroll 2
    for (int k = 0; k < K; k += 32) {
        v16bf a[4], b[4];
#pragma unroll
        for (int i = 0; i < 4; ++i) a[i] = frag_at(Ab[i] + k);
#pragma unroll
        for (int j = 0; j < 4; ++j) b[j] = frag_at(Bb[j] + k);
#pragma unroll
        for (int i = 0; i < 4; ++i)
#pragma unroll
            for (int j = 0; j < 4; ++j)
                acc[i][j] = wmma_bf16(a[i], b[j], acc[i][j]);
    }

    const int cn = lane & 15;
    const int rm = (lane >> 4) << 3;      // D: lanes 16-31 hold M=r+8
#pragma unroll
    for (int j = 0; j < 4; ++j) {
        const int n = n0 + 16 * j + cn;
        if (CLAMP && n >= N) continue;
        const float bn = (bias0 ? bias0[n] : 0.f) + (bias1 ? bias1[n] : 0.f);
#pragma unroll
        for (int i = 0; i < 4; ++i) {
            const int mb = m0 + 16 * i + rm;
#pragma unroll
            for (int r = 0; r < 8; ++r)
                C[(size_t)(mb + r) * N + n] = acc[i][j][r] + bn;
        }
    }
}

// ---------------------------------------------------------------------------
// One LSTM timestep, fully fused:
//   gates[32,4096] = h_prev @ Whh^T + xg_t ; i,f,g,o -> c,h update.
// Each block owns a 32-wide hidden chunk j0..j0+31 and computes ALL FOUR gate
// tiles for it (8 waves = 4 gates x 2 column halves, each wave 32x16 = 2 WMMA
// accumulators over M=batch=32), exchanges through LDS, then 256 threads do
// the pointwise update. grid = H/32 = 32 blocks. Whh (8MB bf16) is L2-resident
// across all 256 steps; this kernel is latency-bound by the recurrence.
// ---------------------------------------------------------------------------
__global__ __launch_bounds__(256) void lstm_step(
    const __bf16* __restrict__ hprev,   // [B, H] bf16
    const __bf16* __restrict__ Whh,     // [4H, H] bf16 row-major
    const float*  __restrict__ xgt,     // [B, 4H] f32 (xg for this t, biases included)
    float*        __restrict__ c_cur,   // [B, H] f32 (in/out)
    float*        __restrict__ h_cur,   // [B, H] f32 (out, final value = h_T)
    __bf16*       __restrict__ hout)    // [B, H] bf16 (hseq slot t)
{
    __shared__ float gsm[4][32][33];    // [gate][batch][j] (+1 pad vs bank conflicts)

    const int wave = threadIdx.x >> 5;
    const int lane = threadIdx.x & 31;
    const int gate = wave >> 1;
    const int half = wave & 1;
    const int j0   = blockIdx.x * 32;
    const int n0   = gate * H_SZ + j0 + half * 16;

    const int rr  = lane & 15;
    const int klo = (lane >> 4) << 3;
    const __bf16* A0 = hprev + (size_t)rr * H_SZ + klo;          // batch rows 0-15
    const __bf16* A1 = hprev + (size_t)(16 + rr) * H_SZ + klo;   // batch rows 16-31
    const __bf16* Bb = Whh + (size_t)(n0 + rr) * H_SZ + klo;

    v8f acc0 = {}, acc1 = {};
#pragma unroll 4
    for (int k = 0; k < H_SZ; k += 32) {
        v16bf a0 = frag_at(A0 + k);
        v16bf a1 = frag_at(A1 + k);
        v16bf b  = frag_at(Bb + k);
        acc0 = wmma_bf16(a0, b, acc0);
        acc1 = wmma_bf16(a1, b, acc1);
    }

    const int cn = lane & 15;
    const int rm = (lane >> 4) << 3;
    const int n  = n0 + cn;
    const int j  = half * 16 + cn;
#pragma unroll
    for (int r = 0; r < 8; ++r) {
        const int mA = rm + r;          // 0..15
        const int mB = mA + 16;         // 16..31
        gsm[gate][mA][j] = acc0[r] + xgt[(size_t)mA * G4H + n];
        gsm[gate][mB][j] = acc1[r] + xgt[(size_t)mB * G4H + n];
    }
    __syncthreads();

#pragma unroll
    for (int p = 0; p < 4; ++p) {
        const int e  = threadIdx.x + p * 256;   // 1024 = 32 batch x 32 j
        const int b  = e >> 5;
        const int jj = e & 31;
        const int jg = j0 + jj;
        float iv = gsm[0][b][jj];
        float fv = gsm[1][b][jj];
        float gv = gsm[2][b][jj];
        float ov = gsm[3][b][jj];
        iv = 1.f / (1.f + __expf(-iv));
        fv = 1.f / (1.f + __expf(-fv));
        gv = tanhf(gv);
        ov = 1.f / (1.f + __expf(-ov));
        const size_t idx = (size_t)b * H_SZ + jg;
        const float cnew = fv * c_cur[idx] + iv * gv;
        const float hnew = ov * tanhf(cnew);
        c_cur[idx] = cnew;
        h_cur[idx] = hnew;
        hout[idx]  = f2bf(hnew);
    }
}

// ---------------------------------------------------------------------------
__global__ void k_f32_to_bf16(const float* __restrict__ src, __bf16* __restrict__ dst, long n) {
    long i = (long)blockIdx.x * blockDim.x + threadIdx.x;
    const long stride = (long)gridDim.x * blockDim.x;
    for (; i < n; i += stride) dst[i] = f2bf(src[i]);
}

__global__ void k_copy_f32(const float* __restrict__ src, float* __restrict__ dst, long n) {
    long i = (long)blockIdx.x * blockDim.x + threadIdx.x;
    const long stride = (long)gridDim.x * blockDim.x;
    for (; i < n; i += stride) dst[i] = src[i];
}

// ---------------------------------------------------------------------------
extern "C" void kernel_launch(void* const* d_in, const int* in_sizes, int n_in,
                              void* d_out, int out_size, void* d_ws, size_t ws_size,
                              hipStream_t stream) {
    (void)in_sizes; (void)n_in; (void)out_size; (void)ws_size;

    const float* x    = (const float*)d_in[0];   // (T,B,H)
    const float* h0   = (const float*)d_in[1];   // (L,B,H)
    const float* c0   = (const float*)d_in[2];   // (L,B,H)
    const float* Wih  = (const float*)d_in[3];   // (L,4H,H)
    const float* Whh  = (const float*)d_in[4];   // (L,4H,H)
    const float* bih  = (const float*)d_in[5];   // (L,4H)
    const float* bhh  = (const float*)d_in[6];   // (L,4H)
    const float* Wdec = (const float*)d_in[7];   // (V,H)
    const float* bdec = (const float*)d_in[8];   // (V,)
    float* out = (float*)d_out;                  // logits | h_n | c_n

    // ---- workspace carve-up (256B aligned) ----
    char* w = (char*)d_ws;
    size_t off = 0;
    auto carve = [&](size_t bytes) -> void* {
        void* p = w + off;
        off += (bytes + 255) & ~(size_t)255;
        return p;
    };
    __bf16* x_bf    = (__bf16*)carve((size_t)T_SEQ * B_SZ * H_SZ * 2);
    __bf16* Wih_bf  = (__bf16*)carve((size_t)L_SZ * G4H * H_SZ * 2);
    __bf16* Whh_bf  = (__bf16*)carve((size_t)L_SZ * G4H * H_SZ * 2);
    __bf16* Wdec_bf = (__bf16*)carve((size_t)V_SZ * H_SZ * 2);
    __bf16* h0_bf   = (__bf16*)carve((size_t)L_SZ * B_SZ * H_SZ * 2);
    __bf16* hseq_bf = (__bf16*)carve((size_t)L_SZ * T_SEQ * B_SZ * H_SZ * 2); // per-layer h sequence
    float*  xg      = (float*)carve((size_t)T_SEQ * B_SZ * G4H * 4);          // reused per layer
    float*  c_cur   = (float*)carve((size_t)L_SZ * B_SZ * H_SZ * 4);
    float*  h_cur   = (float*)carve((size_t)L_SZ * B_SZ * H_SZ * 4);

    const long nBH = (long)B_SZ * H_SZ;

    // ---- one-time (per call) conversions / state init ----
    k_f32_to_bf16<<<1024, 256, 0, stream>>>(x,    x_bf,    (long)T_SEQ * B_SZ * H_SZ);
    k_f32_to_bf16<<<1024, 256, 0, stream>>>(Wih,  Wih_bf,  (long)L_SZ * G4H * H_SZ);
    k_f32_to_bf16<<<1024, 256, 0, stream>>>(Whh,  Whh_bf,  (long)L_SZ * G4H * H_SZ);
    k_f32_to_bf16<<<1024, 256, 0, stream>>>(Wdec, Wdec_bf, (long)V_SZ * H_SZ);
    k_f32_to_bf16<<<64,   256, 0, stream>>>(h0,   h0_bf,   (long)L_SZ * nBH);
    k_copy_f32  <<<64,   256, 0, stream>>>(c0,   c_cur,   (long)L_SZ * nBH);

    // ---- LSTM layers ----
    for (int l = 0; l < L_SZ; ++l) {
        const __bf16* layer_in = (l == 0) ? x_bf : (hseq_bf + (size_t)0 * T_SEQ * nBH);
        __bf16* hseq_l = hseq_bf + (size_t)l * T_SEQ * nBH;

        // xg[T*B, 4H] = layer_in @ Wih_l^T + bih_l + bhh_l   (M=8192, N=4096, K=1024)
        {
            dim3 grid(G4H / 256, (T_SEQ * B_SZ) / 128);
            gemm_bf16_nt<H_SZ, false><<<grid, 256, 0, stream>>>(
                layer_in, Wih_bf + (size_t)l * G4H * H_SZ,
                bih + (size_t)l * G4H, bhh + (size_t)l * G4H,
                xg, T_SEQ * B_SZ, G4H);
        }

        // sequential recurrence: one fused kernel per timestep
        const __bf16* Whh_l = Whh_bf + (size_t)l * G4H * H_SZ;
        for (int t = 0; t < T_SEQ; ++t) {
            const __bf16* hprev = (t == 0) ? (h0_bf + (size_t)l * nBH)
                                           : (hseq_l + (size_t)(t - 1) * nBH);
            lstm_step<<<H_SZ / 32, 256, 0, stream>>>(
                hprev, Whh_l,
                xg + (size_t)t * B_SZ * G4H,
                c_cur + (size_t)l * nBH,
                h_cur + (size_t)l * nBH,
                hseq_l + (size_t)t * nBH);
        }
    }

    // ---- decoder: logits[T*B, V] = hseq(layer1) @ Wdec^T + bdec ----
    {
        dim3 grid((V_SZ + 255) / 256, (T_SEQ * B_SZ) / 128);
        gemm_bf16_nt<H_SZ, true><<<grid, 256, 0, stream>>>(
            hseq_bf + (size_t)1 * T_SEQ * nBH, Wdec_bf,
            bdec, (const float*)nullptr,
            out, T_SEQ * B_SZ, V_SZ);
    }

    // ---- h_n, c_n outputs ----
    float* out_hn = out + (size_t)T_SEQ * B_SZ * V_SZ;
    float* out_cn = out_hn + (size_t)L_SZ * nBH;
    k_copy_f32<<<64, 256, 0, stream>>>(h_cur, out_hn, (long)L_SZ * nBH);
    k_copy_f32<<<64, 256, 0, stream>>>(c_cur, out_cn, (long)L_SZ * nBH);
}
